// JointAttention_70102456205934
// MI455X (gfx1250) — matrix operand order repaired and checked by reading, not compiled, not run on hardware
//
#include <hip/hip_runtime.h>
#include <stdint.h>

// ---------------- CDNA5 WMMA plumbing ----------------
typedef __bf16 bf16_t;
typedef bf16_t v16bf __attribute__((ext_vector_type(16)));
typedef float  v8f   __attribute__((ext_vector_type(8)));

union FragB16 { uint32_t u[8]; v16bf v; };

// Async global->LDS path (gfx1250); guarded so either toolchain compiles.
#if defined(__gfx1250__) && __has_builtin(__builtin_amdgcn_global_load_async_to_lds_b128)
#define HAVE_ASYNC_LDS 1
#endif

#define AS1 __attribute__((address_space(1)))
#define AS3 __attribute__((address_space(3)))

// exact parameter type per hipcc diagnostic: int __attribute__((vector_size(16))) *
typedef int v4i_vs __attribute__((vector_size(16)));

__device__ __forceinline__ void wait_async_then_barrier() {
#if defined(HAVE_ASYNC_LDS)
#if __has_builtin(__builtin_amdgcn_s_wait_asynccnt)
    __builtin_amdgcn_s_wait_asynccnt(0);
#else
    asm volatile("s_wait_asynccnt 0" ::: "memory");
#endif
#endif
    __syncthreads();
}

__device__ __forceinline__ uint16_t f2bf(float f) {
    uint32_t x = __float_as_uint(f);
    x += 0x7FFFu + ((x >> 16) & 1u);       // round-to-nearest-even
    return (uint16_t)(x >> 16);
}
__device__ __forceinline__ uint32_t packbf(float lo, float hi) {
    return (uint32_t)f2bf(lo) | ((uint32_t)f2bf(hi) << 16);
}
__device__ __forceinline__ v8f wmma_bf16(const FragB16& a, const FragB16& b, v8f c) {
    // D = A(16x32 bf16) * B(32x16 bf16) + C(16x16 f32)
    return __builtin_amdgcn_wmma_f32_16x16x32_bf16(false, a.v, false, b.v,
                                                   (short)0, c, false, false);
}

// Problem constants
static constexpr int Bb  = 2;
static constexpr int TNc = 2048;
static constexpr int Hc  = 16;
static constexpr int Dc  = 64;
static constexpr int DIc = 1024;   // H*D

// ---------------- elementwise f32 -> bf16 ----------------
__global__ void k_f32_to_bf16(const float* __restrict__ src,
                              uint16_t* __restrict__ dst, int npairs) {
    int i = blockIdx.x * blockDim.x + threadIdx.x;
    if (i < npairs) {
        float2 v = ((const float2*)src)[i];
        ((uint32_t*)dst)[i] = packbf(v.x, v.y);
    }
}

// ---------------- w_tau transpose-convert: [K][C] f32 -> [C][K] bf16 ----------------
__global__ __launch_bounds__(256)
void k_transpose_bf16(const float* __restrict__ src, uint16_t* __restrict__ dst,
                      int K, int C) {
    __shared__ float t[32][33];
    const int c0 = blockIdx.x * 32, k0 = blockIdx.y * 32;
    for (int e = threadIdx.x; e < 1024; e += 256) {
        int kl = e >> 5, cl = e & 31;
        t[kl][cl] = src[(size_t)(k0 + kl) * C + c0 + cl];
    }
    __syncthreads();
    for (int e = threadIdx.x; e < 1024; e += 256) {
        int kl = e & 31, cl = e >> 5;     // consecutive threads -> consecutive k (coalesced)
        dst[(size_t)(c0 + cl) * K + k0 + kl] = f2bf(t[kl][cl]);
    }
}

// ---------------- generic tiled bf16 GEMM ----------------
// C[M,N] = A[M,K](bf16,row-major via batch row-map) x B[K,N](bf16,row-major)
// EPI=0: f32 row-major output.  EPI=1: QKV scatter epilogue (+ v transpose copy).
template<int EPI>
__global__ __launch_bounds__(256)
void k_gemm_bf16(const uint16_t* __restrict__ A, const uint16_t* __restrict__ Bm,
                 int M, int N, int K, int lda,
                 int rpb, int aoff, int abstride,
                 float* __restrict__ outf,
                 uint16_t* __restrict__ qd, uint16_t* __restrict__ kd,
                 uint16_t* __restrict__ vd, uint16_t* __restrict__ vTd,
                 int Nmod, int noff)
{
    __shared__ uint16_t lsa[128 * 36];   // A tile  [128 rows][32 k] stride 36
    __shared__ uint16_t lsb[128 * 36];   // B tile transposed [128 cols][32 k] stride 36

    const int tid  = threadIdx.x;
    const int lane = tid & 31, wave = tid >> 5;
    const int ln   = lane & 15, lh = lane >> 4;
    const int wm   = wave >> 1, wn = wave & 1;       // 4x2 waves; wave tile 32x64
    const int m0   = blockIdx.y * 128, n0 = blockIdx.x * 128;

    const v8f z8 = {0,0,0,0,0,0,0,0};
    v8f acc[2][4];
    for (int i = 0; i < 2; i++) for (int j = 0; j < 4; j++) acc[i][j] = z8;

    const int kadw = lh ? 8 : 0;   // A-frag dword offset (K 0..15 / 16..31)
    const int kbdw = lh ? 8 : 0;   // B-frag dword offset (K 0..15 / 16..31)

    for (int k0 = 0; k0 < K; k0 += 32) {
        // stage A (dword loads, coalesced)
        for (int e = tid; e < 2048; e += 256) {
            int r = e >> 4, c2 = e & 15;
            int rl = m0 + r;
            int bb = rl / rpb, nn = rl % rpb;
            size_t arow = (size_t)bb * abstride + aoff + nn;
            ((uint32_t*)lsa)[r * 18 + c2] =
                *(const uint32_t*)(A + arow * lda + k0 + 2 * c2);
        }
        // stage B transposed
        for (int e = tid; e < 2048; e += 256) {
            int kk = e >> 6, n2 = e & 63;
            uint32_t d = *(const uint32_t*)(Bm + (size_t)(k0 + kk) * N + n0 + 2 * n2);
            lsb[(2 * n2)     * 36 + kk] = (uint16_t)d;
            lsb[(2 * n2 + 1) * 36 + kk] = (uint16_t)(d >> 16);
        }
        __syncthreads();

        FragB16 af[2], bfr[4];
        #pragma unroll
        for (int tm = 0; tm < 2; tm++) {
            const uint32_t* base = (const uint32_t*)lsa + (wm * 32 + tm * 16 + ln) * 18;
            #pragma unroll
            for (int j = 0; j < 4; j++) {
                af[tm].u[j]     = base[kadw + j];
                af[tm].u[4 + j] = base[4 + kadw + j];
            }
        }
        #pragma unroll
        for (int tn = 0; tn < 4; tn++) {
            const uint32_t* base = (const uint32_t*)lsb + (wn * 64 + tn * 16 + ln) * 18;
            #pragma unroll
            for (int j = 0; j < 8; j++) bfr[tn].u[j] = base[kbdw + j];
        }
        #pragma unroll
        for (int tm = 0; tm < 2; tm++)
            #pragma unroll
            for (int tn = 0; tn < 4; tn++)
                acc[tm][tn] = wmma_bf16(af[tm], bfr[tn], acc[tm][tn]);
        __syncthreads();
    }

    // epilogue
    #pragma unroll
    for (int tm = 0; tm < 2; tm++) {
        #pragma unroll
        for (int tn = 0; tn < 4; tn++) {
            #pragma unroll
            for (int t = 0; t < 8; t++) {
                int mm = lh ? 8 + t : t;
                int m  = m0 + wm * 32 + tm * 16 + mm;
                int c  = n0 + wn * 64 + tn * 16 + ln;
                float val = acc[tm][tn][t];
                if (EPI == 0) {
                    outf[(size_t)m * N + c] = val;
                } else {
                    int which = c >> 10, rem = c & 1023;
                    int hh = rem >> 6, dd = rem & 63;
                    int bb = m / Nmod, nn = m % Nmod;
                    uint16_t bv = f2bf(val);
                    size_t idx = ((size_t)(bb * Hc + hh) * TNc + noff + nn) * Dc + dd;
                    if (which == 0)      qd[idx] = bv;
                    else if (which == 1) kd[idx] = bv;
                    else {
                        vd[idx] = bv;
                        vTd[((size_t)(bb * Hc + hh) * Dc + dd) * TNc + noff + nn] = bv;
                    }
                }
            }
        }
    }
}

// ---------------- fused taus + sim + softmax + PV ----------------
// grid: (TN/16, H, B); 256 threads (8 waves). ~200 KB dynamic LDS.
static constexpr int SIMS = 2052;                 // sim row stride (floats)
static constexpr size_t ATTN_LDS =
    (size_t)16 * SIMS * 4 +        // sim buffer (f32)
    8192 * 4 +                     // 8-wave PV partials (16x64 f32 each)
    (size_t)16 * 1032 * 2 +        // qflat rows (bf16, K=1024, stride 1032)
    (size_t)16 * 72 * 2 +          // q head rows (bf16, D=64, stride 72)
    256 * 4 + 32 * 4;              // reduction scratch

__global__ __launch_bounds__(256)
void k_attention(const uint16_t* __restrict__ qb, const uint16_t* __restrict__ kb,
                 const uint16_t* __restrict__ vTb,
                 const uint16_t* __restrict__ wtauT, const float* __restrict__ btau,
                 const float* __restrict__ attn_dist,
                 const uint8_t* __restrict__ mask0, const uint8_t* __restrict__ mask1,
                 uint16_t* __restrict__ outb)
{
    extern __shared__ char smem[];
    float*    simbuf   = (float*)smem;                       // 16 * SIMS
    float*    partials = simbuf + 16 * SIMS;                 // 8 * 1024
    uint16_t* qflat_s  = (uint16_t*)(partials + 8192);       // 16 * 1032
    uint16_t* qh_s     = qflat_s + 16 * 1032;                // 16 * 72
    float*    red      = (float*)(qh_s + 16 * 72);           // 256
    float*    rowmax   = red + 256;                          // 16
    float*    rowsum   = rowmax + 16;                        // 16

    const int tid = threadIdx.x, lane = tid & 31, wave = tid >> 5;
    const int ln = lane & 15, lh = lane >> 4;
    const int it = blockIdx.x, h = blockIdx.y, b = blockIdx.z;
    const int i0 = it * 16;
    const v8f z8 = {0,0,0,0,0,0,0,0};

    // ---- stage q (flat-reshape rows + this head's rows) into LDS ----
#if defined(HAVE_ASYNC_LDS)
    {
        // qflat: 16 rows x 2048B, LDS row stride 2064B (16B aligned)
        const char* srcb = (const char*)qb +
            ((size_t)b * Hc * TNc * Dc + (size_t)i0 * DIc) * 2;
        char* dstb = (char*)qflat_s;
        for (int e = tid; e < 2048; e += 256) {
            int r = e >> 7, cw = e & 127;
            __builtin_amdgcn_global_load_async_to_lds_b128(
                (AS1 v4i_vs*)(srcb + (size_t)r * 2048 + cw * 16),
                (AS3 v4i_vs*)(dstb + r * 2064 + cw * 16), 0, 0);
        }
        // qh: 16 rows x 128B, LDS row stride 144B (16B aligned)
        const char* srch = (const char*)qb +
            ((size_t)((b * Hc + h) * TNc + i0) * Dc) * 2;
        char* dsth = (char*)qh_s;
        for (int e = tid; e < 128; e += 256) {
            int r = e >> 3, cw = e & 7;
            __builtin_amdgcn_global_load_async_to_lds_b128(
                (AS1 v4i_vs*)(srch + (size_t)r * 128 + cw * 16),
                (AS3 v4i_vs*)(dsth + r * 144 + cw * 16), 0, 0);
        }
    }
#else
    {
        const uint32_t* src =
            (const uint32_t*)qb + ((size_t)b * Hc * TNc * Dc + (size_t)i0 * DIc) / 2;
        for (int e = tid; e < 16 * 512; e += 256) {
            int r = e >> 9, cw = e & 511;
            ((uint32_t*)qflat_s)[r * 516 + cw] = src[r * 512 + cw];
        }
        const uint32_t* srch =
            (const uint32_t*)qb + ((size_t)((b * Hc + h) * TNc + i0) * Dc) / 2;
        for (int e = tid; e < 16 * 32; e += 256) {
            int r = e >> 5, cw = e & 31;
            ((uint32_t*)qh_s)[r * 36 + cw] = srch[r * 32 + cw];
        }
    }
#endif
    wait_async_then_barrier();

    const int kadw = lh ? 8 : 0;
    const int kbdw = lh ? 8 : 0;

    // ---- phase 1: sim/8 - (tau + b_tau)*attn_dist into LDS ----
    for (int jt = wave; jt < 128; jt += 8) {
        const int j0 = jt * 16;

        // sim = q_i (16x64) . k_j^T  — two K=32 steps
        v8f sacc = z8;
        const uint32_t* kg =
            (const uint32_t*)kb + ((size_t)((b * Hc + h) * TNc + j0 + ln) * Dc) / 2;
        #pragma unroll
        for (int ks = 0; ks < 2; ks++) {
            FragB16 a, bb_;
            const uint32_t* ab = (const uint32_t*)qh_s + ln * 36 + ks * 16;
            #pragma unroll
            for (int j = 0; j < 4; j++) { a.u[j] = ab[kadw + j]; a.u[4 + j] = ab[4 + kadw + j]; }
            const uint32_t* bbp = kg + ks * 16 + kbdw;
            #pragma unroll
            for (int j = 0; j < 8; j++) bb_.u[j] = bbp[j];
            sacc = wmma_bf16(a, bb_, sacc);
        }

        // tau tile: qflat_i (16x1024) . w_tau cols (j0+n)*16+h  — 32 K-steps,
        // two-deep software pipeline on the streamed B fragments (L2-resident).
        v8f tacc = z8;
        const int ccol = (j0 + ln) * Hc + h;
        const uint32_t* wg  = (const uint32_t*)wtauT + (size_t)ccol * 512 + kbdw;
        const uint32_t* qfl = (const uint32_t*)qflat_s + ln * 516;
        FragB16 bcur, bnxt;
        #pragma unroll
        for (int j = 0; j < 8; j++) bcur.u[j] = wg[j];
        for (int kk = 0; kk < 1024; kk += 32) {
            if (kk < 1024 - 32) {
                const uint32_t* nb = wg + ((kk + 32) >> 1);
                __builtin_prefetch(nb + 16, 0, 1);      // two K-steps ahead
                #pragma unroll
                for (int j = 0; j < 8; j++) bnxt.u[j] = nb[j];
            }
            FragB16 a;
            const uint32_t* ab = qfl + (kk >> 1);
            #pragma unroll
            for (int j = 0; j < 4; j++) { a.u[j] = ab[kadw + j]; a.u[4 + j] = ab[4 + kadw + j]; }
            tacc = wmma_bf16(a, bcur, tacc);
            bcur = bnxt;
        }

        // epilogue: scale, mask, bias, scatter to sim buffer (C-layout: col=ln, rows=t)
        const int j = j0 + ln;
        const float btc = btau[ccol];
        const uint8_t mk = (j < 1024) ? mask0[b * 1024 + j] : mask1[b * 1024 + (j - 1024)];
        #pragma unroll
        for (int t = 0; t < 8; t++) {
            int m = lh ? 8 + t : t;
            int i = i0 + m;
            float ad = attn_dist[((size_t)b * TNc + i) * TNc + j];
            float s  = mk ? sacc[t] * 0.125f : -3.402823466e38f;
            s -= (tacc[t] + btc) * ad;
            simbuf[m * SIMS + j] = s;
        }
    }
    __syncthreads();

    // ---- phase 2: row softmax over 2048 cols (f32) ----
    {
        const int r = tid >> 4, part = tid & 15;
        float* row = simbuf + r * SIMS;
        float mx = -3.402823466e38f;
        for (int c = part; c < TNc; c += 16) mx = fmaxf(mx, row[c]);
        red[r * 16 + part] = mx;
        __syncthreads();
        if (part == 0) {
            float m2 = -3.402823466e38f;
            for (int p = 0; p < 16; p++) m2 = fmaxf(m2, red[r * 16 + p]);
            rowmax[r] = m2;
        }
        __syncthreads();
        const float rm = rowmax[r];
        float sm = 0.f;
        for (int c = part; c < TNc; c += 16) sm += __expf(row[c] - rm);
        red[r * 16 + part] = sm;
        __syncthreads();
        if (part == 0) {
            float s2 = 0.f;
            for (int p = 0; p < 16; p++) s2 += red[r * 16 + p];
            rowsum[r] = s2;
        }
        __syncthreads();
        const float inv = 1.0f / rowsum[r];
        for (int c = part; c < TNc; c += 16) row[c] = __expf(row[c] - rm) * inv;
    }
    __syncthreads();

    // ---- phase 3: out(16x64) = P(16x2048) . V ; split-K over 8 waves ----
    {
        v8f oacc[4];
        for (int i = 0; i < 4; i++) oacc[i] = z8;
        const int jb = wave * 256;
        for (int ks = 0; ks < 8; ks++) {
            const int kj = jb + ks * 32;
            FragB16 a;
            const float* prow = simbuf + ln * SIMS + kj;
            const int ko = lh ? 16 : 0;
            #pragma unroll
            for (int j = 0; j < 4; j++) {
                a.u[j]     = packbf(prow[ko + 2 * j],      prow[ko + 2 * j + 1]);
                a.u[4 + j] = packbf(prow[8 + ko + 2 * j], prow[8 + ko + 2 * j + 1]);
            }
            #pragma unroll
            for (int dt = 0; dt < 4; dt++) {
                FragB16 bb_;
                const int d = dt * 16 + ln;
                const uint32_t* vb = (const uint32_t*)vTb +
                    (((size_t)(b * Hc + h) * Dc + d) * TNc + kj) / 2 + kbdw;
                #pragma unroll
                for (int j = 0; j < 8; j++) bb_.u[j] = vb[j];
                oacc[dt] = wmma_bf16(a, bb_, oacc[dt]);
            }
        }
        #pragma unroll
        for (int dt = 0; dt < 4; dt++)
            #pragma unroll
            for (int t = 0; t < 8; t++)
                partials[wave * 1024 + dt * 256 + t * 32 + lane] = oacc[dt][t];
    }
    __syncthreads();

    // cross-wave reduction + store to (B, TN, H*D) bf16
    for (int e = tid; e < 1024; e += 256) {
        float s = 0.f;
        #pragma unroll
        for (int w = 0; w < 8; w++) s += partials[w * 1024 + e];
        int dt = e >> 8, rem = e & 255, v = rem >> 5, l2 = rem & 31;
        int m = (l2 < 16) ? v : 8 + v;
        int d = dt * 16 + (l2 & 15);
        outb[((size_t)(b * TNc + i0 + m)) * DIc + h * Dc + d] = f2bf(s);
    }
}

// ---------------- host launcher ----------------
extern "C" void kernel_launch(void* const* d_in, const int* in_sizes, int n_in,
                              void* d_out, int out_size, void* d_ws, size_t ws_size,
                              hipStream_t stream)
{
    (void)in_sizes; (void)n_in; (void)out_size; (void)ws_size;

    const float*   x0        = (const float*)d_in[0];
    const float*   x1        = (const float*)d_in[1];
    const float*   attn_dist = (const float*)d_in[2];
    const uint8_t* mask0     = (const uint8_t*)d_in[3];
    const uint8_t* mask1     = (const uint8_t*)d_in[4];
    const float*   w_qkv0    = (const float*)d_in[5];
    const float*   w_qkv1    = (const float*)d_in[6];
    const float*   w_out0    = (const float*)d_in[7];
    const float*   w_out1    = (const float*)d_in[8];
    const float*   w_tau     = (const float*)d_in[9];
    const float*   b_tau     = (const float*)d_in[10];

    char* ws = (char*)d_ws;
    size_t off = 0;
    auto carve = [&](size_t bytes) -> char* {
        char* p = ws + off; off += (bytes + 255) & ~(size_t)255; return p;
    };
    uint16_t* xb0    = (uint16_t*)carve((size_t)2 * 1024 * 1024 * 2);
    uint16_t* xb1    = (uint16_t*)carve((size_t)2 * 1024 * 768 * 2);
    uint16_t* wqkv0b = (uint16_t*)carve((size_t)1024 * 3072 * 2);
    uint16_t* wqkv1b = (uint16_t*)carve((size_t)768 * 3072 * 2);
    uint16_t* wout0b = (uint16_t*)carve((size_t)1024 * 1024 * 2);
    uint16_t* wout1b = (uint16_t*)carve((size_t)1024 * 768 * 2);
    uint16_t* wtauT  = (uint16_t*)carve((size_t)32768 * 1024 * 2);
    uint16_t* qbuf   = (uint16_t*)carve((size_t)2 * 16 * 2048 * 64 * 2);
    uint16_t* kbuf   = (uint16_t*)carve((size_t)2 * 16 * 2048 * 64 * 2);
    uint16_t* vbuf   = (uint16_t*)carve((size_t)2 * 16 * 2048 * 64 * 2);
    uint16_t* vTbuf  = (uint16_t*)carve((size_t)2 * 16 * 64 * 2048 * 2);
    uint16_t* attnout= (uint16_t*)carve((size_t)2 * 2048 * 1024 * 2);

    auto conv = [&](const float* s, uint16_t* d, int pairs) {
        k_f32_to_bf16<<<(pairs + 255) / 256, 256, 0, stream>>>(s, d, pairs);
    };
    conv(x0,     xb0,    2 * 1024 * 1024 / 2);
    conv(x1,     xb1,    2 * 1024 * 768  / 2);
    conv(w_qkv0, wqkv0b, 1024 * 3072 / 2);
    conv(w_qkv1, wqkv1b, 768  * 3072 / 2);
    conv(w_out0, wout0b, 1024 * 1024 / 2);
    conv(w_out1, wout1b, 1024 * 768  / 2);
    k_transpose_bf16<<<dim3(32768 / 32, 1024 / 32), 256, 0, stream>>>(w_tau, wtauT, 1024, 32768);

    // QKV projections: M = B*Nmod = 2048, N = 3072
    k_gemm_bf16<1><<<dim3(3072 / 128, 2048 / 128), 256, 0, stream>>>(
        xb0, wqkv0b, 2048, 3072, 1024, 1024,
        2048, 0, 2048, nullptr, qbuf, kbuf, vbuf, vTbuf, 1024, 0);
    k_gemm_bf16<1><<<dim3(3072 / 128, 2048 / 128), 256, 0, stream>>>(
        xb1, wqkv1b, 2048, 3072, 768, 768,
        2048, 0, 2048, nullptr, qbuf, kbuf, vbuf, vTbuf, 1024, 1024);

    // fused taus + attention
    (void)hipFuncSetAttribute((const void*)k_attention,
                              hipFuncAttributeMaxDynamicSharedMemorySize, (int)ATTN_LDS);
    k_attention<<<dim3(2048 / 16, 16, 2), 256, ATTN_LDS, stream>>>(
        qbuf, kbuf, vTbuf, wtauT, b_tau, attn_dist, mask0, mask1, attnout);

    // output projections -> concatenated f32 d_out
    float* out0 = (float*)d_out;
    float* out1 = out0 + (size_t)2 * 1024 * 1024;
    k_gemm_bf16<0><<<dim3(1024 / 128, 2048 / 128), 256, 0, stream>>>(
        attnout, wout0b, 2048, 1024, 1024, 1024,
        1024, 0, 2048, out0, nullptr, nullptr, nullptr, nullptr, 0, 0);
    k_gemm_bf16<0><<<dim3(768 / 128, 2048 / 128), 256, 0, stream>>>(
        attnout, wout1b, 2048, 768, 1024, 1024,
        1024, 1024, 2048, out1, nullptr, nullptr, nullptr, nullptr, 0, 0);
}
